// GCN_Layer_37933151158789
// MI455X (gfx1250) — compile-verified, matrix operand order checked by reading, not run on hardware
//
#include <hip/hip_runtime.h>
#include <cstdint>
#include <cstddef>

#define NBATCH 8
#define NNODE  2048
#define FDIM   512

typedef __attribute__((ext_vector_type(16))) __bf16    v16bf;
typedef __attribute__((ext_vector_type(8)))  __bf16    v8bf;
typedef __attribute__((ext_vector_type(16))) float     v16f;
typedef __attribute__((ext_vector_type(8)))  float     v8f;

// ---- fragment builders -----------------------------------------------------
// A/B fragment 16-bit layout (§7.12.2): elements 0..7 <- p[0..7],
// elements 8..15 <- p[16..23] (k-chunks per lane group selected by caller).

// from f32 memory: four 16B loads + 8x v_cvt_pk_bf16_f32
__device__ __forceinline__ v16bf frag_from_f32(const float* __restrict__ p) {
    struct Q { float4 a, b, c, d; } q;
    q.a = *(const float4*)(p + 0);
    q.b = *(const float4*)(p + 4);
    q.c = *(const float4*)(p + 16);
    q.d = *(const float4*)(p + 20);
    v16f f = __builtin_bit_cast(v16f, q);
    return __builtin_convertvector(f, v16bf);
}

// from bf16 memory: two 16B loads, no conversion
__device__ __forceinline__ v16bf frag_from_bf16(const unsigned short* __restrict__ p) {
    struct Pair { uint4 a, b; } pr;
    pr.a = *(const uint4*)(p + 0);
    pr.b = *(const uint4*)(p + 16);
    return __builtin_bit_cast(v16bf, pr);
}

// ============================================================================
// Kernel 0a: Wt[n][k] = bf16(W[k][n])   (one-shot 0.5MB transpose+convert)
// ============================================================================
__global__ void gcn_wcvt_kernel(const float* __restrict__ W,
                                unsigned short* __restrict__ Wt) {
    const int idx = (int)(blockIdx.x * blockDim.x + threadIdx.x);
    if (idx >= FDIM * FDIM) return;
    const int k = idx / FDIM;
    const int n = idx % FDIM;                       // coalesced read over n
    __bf16 v = (__bf16)W[idx];
    Wt[(size_t)n * FDIM + k] = __builtin_bit_cast(unsigned short, v);
}

// ============================================================================
// Kernel 0b: Xbf = bf16(X)  elementwise, 8 floats per thread
// ============================================================================
__global__ void gcn_xcvt_kernel(const float* __restrict__ X,
                                unsigned short* __restrict__ Xbf) {
    const size_t idx = (size_t)(blockIdx.x * blockDim.x + threadIdx.x);
    const size_t total = (size_t)NBATCH * NNODE * FDIM / 8;
    if (idx >= total) return;
    struct Q { float4 a, b; } q;
    q.a = *(const float4*)(X + idx * 8);
    q.b = *(const float4*)(X + idx * 8 + 4);
    v8f f = __builtin_bit_cast(v8f, q);
    v8bf c = __builtin_convertvector(f, v8bf);
    *(uint4*)(Xbf + idx * 8) = __builtin_bit_cast(uint4, c);
}

// ============================================================================
// Kernel 1: dinv[b*N+i] = rsqrt(1 + sum_j A[b,i,j])   (one wave32 per row)
// ============================================================================
__global__ void gcn_degree_kernel(const float* __restrict__ A,
                                  float* __restrict__ dinv) {
    const int wid  = (int)((blockIdx.x * blockDim.x + threadIdx.x) >> 5);
    const int lane = (int)(threadIdx.x & 31);
    if (wid >= NBATCH * NNODE) return;

    const float* row = A + (size_t)wid * NNODE;
    float s = 0.0f;
    for (int j = lane * 4; j < NNODE; j += 32 * 4) {
        float4 v = *(const float4*)(row + j);
        s += v.x + v.y + v.z + v.w;
    }
    for (int off = 16; off > 0; off >>= 1)
        s += __shfl_down(s, off, 32);
    if (lane == 0)
        dinv[wid] = rsqrtf(s + 1.0f);
}

// ============================================================================
// Kernel 2: xs_t[b][f][j] = bf16( dinv[b,j] * ((X@W)[b,j,f] + bias[f]) )
//   Wave tile 16x64. A from bf16 Xbf, B from bf16 Wt: inner loop is pure
//   b128 loads + WMMA. Output stored TRANSPOSED (f-major) -> 128-bit stores.
// ============================================================================
__global__ void gcn_xw_kernel(const unsigned short* __restrict__ Xbf,
                              const unsigned short* __restrict__ Wt,
                              const float* __restrict__ bias,
                              const float* __restrict__ dinv,
                              unsigned short* __restrict__ xs_t) {
    const int lane = (int)(threadIdx.x & 31);
    const int wid  = (int)((blockIdx.x * blockDim.x + threadIdx.x) >> 5);
    const int fGroups = FDIM / 64;                       // 8
    const int nJobs   = (NBATCH * NNODE / 16) * fGroups; // 8192
    if (wid >= nJobs) return;

    const int mBase = (wid / fGroups) * 16;              // global row (b*N+node)
    const int fBase = (wid % fGroups) * 64;
    const int hi    = lane >> 4;
    const int l16   = lane & 15;

    v8f acc[4] = {};

    for (int k0 = 0; k0 < FDIM; k0 += 32) {
        v16bf a = frag_from_bf16(Xbf + (size_t)(mBase + l16) * FDIM + k0 + hi * 8);
        #pragma unroll
        for (int t = 0; t < 4; ++t) {
            const int n = fBase + t * 16 + l16;
            v16bf b = frag_from_bf16(Wt + (size_t)n * FDIM + k0 + hi * 8);
            acc[t] = __builtin_amdgcn_wmma_f32_16x16x32_bf16(
                false, a, false, b, (short)0, acc[t], false, false);
        }
    }

    // epilogue: + bias, * dinv[row], 8 consecutive rows -> one 16B store
    const int batch = mBase / NNODE;
    const int node0 = (mBase % NNODE) + hi * 8;
    float dv[8];
    #pragma unroll
    for (int r = 0; r < 8; ++r) dv[r] = dinv[mBase + hi * 8 + r];

    #pragma unroll
    for (int t = 0; t < 4; ++t) {
        const int col = fBase + t * 16 + l16;
        const float bv = bias[col];
        v8f vv;
        #pragma unroll
        for (int r = 0; r < 8; ++r) vv[r] = (acc[t][r] + bv) * dv[r];
        v8bf c = __builtin_convertvector(vv, v8bf);
        *(uint4*)(xs_t + ((size_t)batch * FDIM + col) * NNODE + node0) =
            __builtin_bit_cast(uint4, c);
    }
}

// ============================================================================
// Kernel 3: out[b,i,f] = dinv[b,i] * ( (A @ xs)[b,i,f] + xs[b,i,f] )
//   Wave tile 16x128 (8 accums): A stream re-read only 4x from L2.
//   A: f32 b128 loads + v_cvt_pk_bf16_f32.  B: K-contiguous bf16 from xs_t.
// ============================================================================
__global__ void gcn_agg_kernel(const float* __restrict__ A,
                               const unsigned short* __restrict__ xs_t,
                               const float* __restrict__ dinv,
                               float* __restrict__ out) {
    const int lane = (int)(threadIdx.x & 31);
    const int wid  = (int)((blockIdx.x * blockDim.x + threadIdx.x) >> 5);
    const int fGroups      = FDIM / 128;                 // 4
    const int jobsPerBatch = (NNODE / 16) * fGroups;     // 512
    if (wid >= NBATCH * jobsPerBatch) return;

    const int batch = wid / jobsPerBatch;
    const int rem   = wid % jobsPerBatch;
    const int iBase = (rem / fGroups) * 16;
    const int fBase = (rem % fGroups) * 128;
    const int hi    = lane >> 4;
    const int l16   = lane & 15;

    const float* Ab          = A    + (size_t)batch * NNODE * NNODE;
    const unsigned short* xb = xs_t + (size_t)batch * FDIM * NNODE;

    v8f acc[8] = {};

    for (int k0 = 0; k0 < NNODE; k0 += 32) {
        v16bf a = frag_from_f32(Ab + (size_t)(iBase + l16) * NNODE + k0 + hi * 8);
        #pragma unroll
        for (int t = 0; t < 8; ++t) {
            const int n = fBase + t * 16 + l16;
            v16bf b = frag_from_bf16(xb + (size_t)n * NNODE + k0 + hi * 8);
            acc[t] = __builtin_amdgcn_wmma_f32_16x16x32_bf16(
                false, a, false, b, (short)0, acc[t], false, false);
        }
    }

    // epilogue: + self-loop xs[i,f], * dinv[i]
    float dv[8];
    #pragma unroll
    for (int r = 0; r < 8; ++r)
        dv[r] = dinv[batch * NNODE + iBase + hi * 8 + r];

    #pragma unroll
    for (int t = 0; t < 8; ++t) {
        const int col = fBase + t * 16 + l16;
        uint4 sv = *(const uint4*)(xb + (size_t)col * NNODE + iBase + hi * 8);
        v8bf sb  = __builtin_bit_cast(v8bf, sv);
        v8f  self = __builtin_convertvector(sb, v8f);
        #pragma unroll
        for (int r = 0; r < 8; ++r) {
            const int g = batch * NNODE + iBase + hi * 8 + r;
            out[(size_t)g * FDIM + col] = (acc[t][r] + self[r]) * dv[r];
        }
    }
}

// ============================================================================
extern "C" void kernel_launch(void* const* d_in, const int* in_sizes, int n_in,
                              void* d_out, int out_size, void* d_ws, size_t ws_size,
                              hipStream_t stream) {
    (void)in_sizes; (void)n_in; (void)out_size; (void)ws_size;
    const float* X    = (const float*)d_in[0];   // [B, N, F_in]
    const float* A    = (const float*)d_in[1];   // [B, N, N]
    const float* W    = (const float*)d_in[2];   // [F_in, F_out]
    const float* bias = (const float*)d_in[3];   // [F_out]
    float* out = (float*)d_out;                  // [B, N, F_out]

    // workspace: xs_t bf16 [B][F][N] | Xbf bf16 [B*N][F] | Wt bf16 [F][F] | dinv f32
    unsigned short* xs_t = (unsigned short*)d_ws;
    unsigned short* Xbf  = xs_t + (size_t)NBATCH * FDIM * NNODE;
    unsigned short* Wt   = Xbf  + (size_t)NBATCH * NNODE * FDIM;
    float* dinv = (float*)(Wt + (size_t)FDIM * FDIM);

    {   // W transpose+convert
        dim3 grid((FDIM * FDIM + 255) / 256), block(256);
        gcn_wcvt_kernel<<<grid, block, 0, stream>>>(W, Wt);
    }
    {   // X convert: 1,048,576 threads (8 floats each)
        const size_t total = (size_t)NBATCH * NNODE * FDIM / 8;
        dim3 grid((unsigned)((total + 255) / 256)), block(256);
        gcn_xcvt_kernel<<<grid, block, 0, stream>>>(X, Xbf);
    }
    {   // degrees: 16384 rows, one wave each
        dim3 grid((NBATCH * NNODE + 7) / 8), block(256);
        gcn_degree_kernel<<<grid, block, 0, stream>>>(A, dinv);
    }
    {   // X@W: 8192 wave jobs
        const int jobs = (NBATCH * NNODE / 16) * (FDIM / 64);
        dim3 grid((jobs + 7) / 8), block(256);
        gcn_xw_kernel<<<grid, block, 0, stream>>>(Xbf, Wt, bias, dinv, xs_t);
    }
    {   // aggregation: 4096 wave jobs
        const int jobs = NBATCH * (NNODE / 16) * (FDIM / 128);
        dim3 grid((jobs + 7) / 8), block(256);
        gcn_agg_kernel<<<grid, block, 0, stream>>>(A, xs_t, dinv, out);
    }
}
